// DLSTMCell_7816840479037
// MI455X (gfx1250) — matrix operand order repaired and checked by reading, not compiled, not run on hardware
//
#include <hip/hip_runtime.h>

typedef __attribute__((ext_vector_type(16))) _Float16 v16h;
typedef __attribute__((ext_vector_type(8)))  _Float16 v8h;
typedef __attribute__((ext_vector_type(8)))  float    v8f;

#define N_NODES   512
#define B_SZ      128
#define RU        64
#define MEM_DIM   16
#define BOT_DIM   4
#define IN_SZ     66          // 2 input + 64 state
#define OUT_SZ    256         // 4 gates * RU
#define W3_COLS   16896       // IN_SZ*OUT_SZ
#define NRU       32768       // N_NODES*RU
#define BNRU      4194304     // B_SZ*NRU

__device__ __forceinline__ float fsig(float x) {
    return __builtin_amdgcn_rcpf(1.0f + __expf(-x));
}
__device__ __forceinline__ float ftanh(float x) {
    float e = __expf(2.0f * x);
    return (e - 1.0f) * __builtin_amdgcn_rcpf(e + 1.0f);
}

// ---------------------------------------------------------------------------
// Single fused kernel, one block per node (512 blocks x 256 threads = 8 wave32):
//   Phase 0: hypernetwork bottleneck  bot[4] = tanh(tanh(mem@w1+b1)@w2+b2)
//   Phase 1: node weights W[K=96 (pad), 256] computed in f16 straight into LDS
//            in WMMA B-fragment order (never touches HBM).
//   Phase 2: A = concat(input, hx) in f16 into LDS in WMMA A-fragment order.
//   Phase 3: [128x96]x[96x256] GEMM via v_wmma_f32_16x16x32_f16 with fused
//            LSTM gate epilogue writing hy/cy.
// LDS: 49.2KB (W) + 24.6KB (A) -> 4 blocks per WGP (320KB).
// ---------------------------------------------------------------------------
__global__ __launch_bounds__(256) void dlstm_fused_kernel(
    const float* __restrict__ inp, const float* __restrict__ hx,
    const float* __restrict__ cx,
    const float* __restrict__ memory, const float* __restrict__ w1,
    const float* __restrict__ b1,     const float* __restrict__ w2,
    const float* __restrict__ b2,     const float* __restrict__ w3,
    const float* __restrict__ b3,     const float* __restrict__ b_out,
    float* __restrict__ out) {
    const int n    = blockIdx.x;
    const int tid  = threadIdx.x;
    const int wave = tid >> 5;
    const int lane = tid & 31;

    __shared__ __align__(32) _Float16 sW[3 * 16 * 32 * 16];  // 49.2 KB, B-frag order
    __shared__ __align__(32) _Float16 sA[8 * 3 * 32 * 16];   // 24.6 KB, A-frag order
    __shared__ float sM[MEM_DIM];
    __shared__ float sB[BOT_DIM];

    // ---- Phase 0: hypernetwork bottleneck (tiny) ----
    if (tid < MEM_DIM) {
        float a = b1[tid];
        #pragma unroll
        for (int k = 0; k < MEM_DIM; ++k)
            a += memory[n * MEM_DIM + k] * w1[k * MEM_DIM + tid];
        sM[tid] = ftanh(a);
    }
    __syncthreads();
    if (tid < BOT_DIM) {
        float a = b2[tid];
        #pragma unroll
        for (int k = 0; k < MEM_DIM; ++k)
            a += sM[k] * w2[k * BOT_DIM + tid];
        sB[tid] = ftanh(a);
    }
    __syncthreads();

    const float c0 = sB[0], c1 = sB[1], c2 = sB[2], c3 = sB[3];

    // ---- Phase 1: W -> LDS in B-fragment swizzle ----
    // Slot mapping chosen for coalescing: 32 consecutive threads share the same
    // K set and read 32 consecutive w3/b3 columns (128B/wave per source row).
    // B-frag layout: half ((kt*16+ntl)*32+lane)*16 + kk  holds
    //   W[K = kt*32 + (lane>=16?16:0) + kk][col = ntl*16 + lane%16];  K>=66 -> 0.
    #pragma unroll
    for (int it = 0; it < 12; ++it) {
        int s    = it * 256 + tid;       // 0..3071
        int col  = s & 255;              // output column 0..255
        int q    = s >> 8;               // 0..11
        int oct  = q & 1;                // which 8 halves of the fragment
        int khi  = (q >> 1) & 1;         // lane>=16 half of the K tile
        int kt   = q >> 2;               // 0..2
        int lane5 = khi * 16 + (col & 15);
        int ntl  = col >> 4;
        int kb   = kt * 32 + khi * 16 + oct * 8;
        v8h h;
        #pragma unroll
        for (int j = 0; j < 8; ++j) {
            int K = kb + j;
            float v = 0.0f;
            if (K < IN_SZ) {
                int idx = K * OUT_SZ + col;
                v = b3[idx] + c0 * w3[idx] + c1 * w3[W3_COLS + idx] +
                    c2 * w3[2 * W3_COLS + idx] + c3 * w3[3 * W3_COLS + idx];
            }
            h[j] = (_Float16)v;
        }
        ((v8h*)sW)[((kt * 16 + ntl) * 32 + lane5) * 2 + oct] = h;
    }

    // ---- Phase 2: A = concat(inputs[:, n*2:+2], hx[:, n*64:+64]) -> LDS ----
    // A-frag layout: K_local = (lane>=16?16:0) + oct*8 + j.  Slot = (row, octet)
    // so each thread reads 8 consecutive source floats per slot.
    #pragma unroll
    for (int u = 0; u < 6; ++u) {
        int t      = u * 256 + tid;      // 0..1535
        int r      = t / 12;             // batch row 0..127
        int o      = t % 12;             // K octet: K = o*8 + j
        int kt     = o >> 2;
        int lanehi = (o >> 1) & 1;
        int oct    = o & 1;
        int mt     = r >> 4;
        int lane5  = lanehi * 16 + (r & 15);
        v8h h;
        #pragma unroll
        for (int j = 0; j < 8; ++j) {
            int K = o * 8 + j;
            float v = 0.0f;
            if (K < 2)          v = inp[(size_t)r * 1024 + n * 2 + K];
            else if (K < IN_SZ) v = hx[(size_t)r * NRU + n * RU + (K - 2)];
            h[j] = (_Float16)v;
        }
        ((v8h*)sA)[((mt * 3 + kt) * 32 + lane5) * 2 + oct] = h;
    }
    __syncthreads();

    // ---- Phase 3: WMMA GEMM + fused LSTM epilogue ----
    v16h afrag[3];
    #pragma unroll
    for (int kt = 0; kt < 3; ++kt)
        afrag[kt] = *(const v16h*)&sA[((wave * 3 + kt) * 32 + lane) * 16];

    const int colj  = lane & 15;
    const int rbase = wave * 16 + ((lane >> 4) << 3);  // C layout: lanes>=16 -> M+8
    float* outHy = out;
    float* outCy = out + BNRU;

    #pragma unroll
    for (int g4 = 0; g4 < 4; ++g4) {
        v8f acc0 = {}, acc1 = {}, acc2 = {}, acc3 = {};
        #pragma unroll
        for (int kt = 0; kt < 3; ++kt) {
            const _Float16* wb = &sW[kt * 16 * 32 * 16];
            // gate gi lives at N-tile ntl = gi*4 + g4 (gates are 64 cols apart)
            v16h bf0 = *(const v16h*)&wb[((0 * 4 + g4) * 32 + lane) * 16];
            v16h bf1 = *(const v16h*)&wb[((1 * 4 + g4) * 32 + lane) * 16];
            v16h bf2 = *(const v16h*)&wb[((2 * 4 + g4) * 32 + lane) * 16];
            v16h bf3 = *(const v16h*)&wb[((3 * 4 + g4) * 32 + lane) * 16];
            acc0 = __builtin_amdgcn_wmma_f32_16x16x32_f16(false, afrag[kt], false, bf0,
                                                          (short)0, acc0, false, false);
            acc1 = __builtin_amdgcn_wmma_f32_16x16x32_f16(false, afrag[kt], false, bf1,
                                                          (short)0, acc1, false, false);
            acc2 = __builtin_amdgcn_wmma_f32_16x16x32_f16(false, afrag[kt], false, bf2,
                                                          (short)0, acc2, false, false);
            acc3 = __builtin_amdgcn_wmma_f32_16x16x32_f16(false, afrag[kt], false, bf3,
                                                          (short)0, acc3, false, false);
        }
        const int j = g4 * 16 + colj;             // unit index within node (0..63)
        const float boi = b_out[j];
        const float bof = b_out[64 + j];
        const float bog = b_out[128 + j];
        const float boo = b_out[192 + j];
        #pragma unroll
        for (int v = 0; v < 8; ++v) {
            int r = rbase + v;
            size_t idx = (size_t)r * NRU + n * RU + j;
            float i_t = fsig(fsig(acc0[v]) + boi);
            float f_t = fsig(fsig(acc1[v]) + bof);
            float g_t = ftanh(fsig(acc2[v]) + bog);
            float o_t = fsig(fsig(acc3[v]) + boo);
            float c  = cx[idx];
            float cy = c * f_t + i_t * g_t;
            float hy = o_t * ftanh(cy);
            outHy[idx] = hy;
            outCy[idx] = cy;
        }
    }
}

extern "C" void kernel_launch(void* const* d_in, const int* in_sizes, int n_in,
                              void* d_out, int out_size, void* d_ws, size_t ws_size,
                              hipStream_t stream) {
    const float* inputs = (const float*)d_in[0];
    const float* hx     = (const float*)d_in[1];
    const float* cx     = (const float*)d_in[2];
    const float* memory = (const float*)d_in[3];
    const float* w1     = (const float*)d_in[4];
    const float* b1     = (const float*)d_in[5];
    const float* w2     = (const float*)d_in[6];
    const float* b2     = (const float*)d_in[7];
    const float* w3     = (const float*)d_in[8];
    const float* b3     = (const float*)d_in[9];
    const float* b_out  = (const float*)d_in[10];

    (void)d_ws; (void)ws_size; (void)in_sizes; (void)n_in; (void)out_size;

    dlstm_fused_kernel<<<N_NODES, 256, 0, stream>>>(
        inputs, hx, cx, memory, w1, b1, w2, b2, w3, b3, b_out, (float*)d_out);
}
